// MultiHeadAttention_57870389346551
// MI455X (gfx1250) — matrix operand order repaired and checked by reading, not compiled, not run on hardware
//
#include <hip/hip_runtime.h>
#include <hip/hip_bf16.h>

typedef __bf16 bf16_t;
typedef __attribute__((ext_vector_type(16))) __bf16 v16bf;
typedef __attribute__((ext_vector_type(8)))  __bf16 v8bf;
typedef __attribute__((ext_vector_type(4)))  __bf16 v4bf;
typedef __attribute__((ext_vector_type(8)))  float  v8f;
typedef __attribute__((ext_vector_type(4)))  unsigned int v4u;
typedef __attribute__((ext_vector_type(4)))  int v4i;
typedef __attribute__((ext_vector_type(8)))  int v8i;

constexpr int BATCH = 2, SEQ = 2048, DMODEL = 1024, NHEAD = 16, DHEAD = 64;
constexpr int MROWS = BATCH * SEQ;           // 4096
constexpr int BM = 128, BN = 128, BK = 32;   // GEMM block tile
constexpr int LDPAD = 8;                     // LDS row pad (bf16 elems)
constexpr int NKT = SEQ / 32;                // key tiles of 32

// ---- WMMA fragment helpers (layouts per CDNA5 ISA 7.12.2) -------------------

static __device__ __forceinline__ v16bf frag_cat(v8bf lo, v8bf hi) {
  return __builtin_shufflevector(lo, hi, 0,1,2,3,4,5,6,7,8,9,10,11,12,13,14,15);
}
// A 16x32 (16-bit): lanes 0-15 K {0..7,16..23}, lanes 16-31 K {8..15,24..31}
static __device__ __forceinline__ v16bf load_a_frag(const bf16_t* row, int half) {
  const int kb = half * 8;
  v8bf lo = *(const v8bf*)(row + kb);
  v8bf hi = *(const v8bf*)(row + kb + 16);
  return frag_cat(lo, hi);
}
// B 32x16 (16-bit): lanes 0-15 K {0..7,16..23}, lanes 16-31 K {8..15,24..31}
static __device__ __forceinline__ v16bf load_b_frag(const bf16_t* col, int half) {
  const int kb = half * 8;
  v8bf lo = *(const v8bf*)(col + kb);
  v8bf hi = *(const v8bf*)(col + kb + 16);
  return frag_cat(lo, hi);
}
static __device__ __forceinline__ v8f wmma_bf16(v16bf a, v16bf b, v8f c) {
  return __builtin_amdgcn_wmma_f32_16x16x32_bf16(false, a, false, b, (short)0, c,
                                                 false, false);
}

// ---- TDM: 2-D tile load, global -> LDS, with LDS row padding ----------------
// Descriptor per CDNA5 ISA 8.3/8.4. Tensor == tile (always in-bounds), bf16
// elements (data_size=1 -> 2B). pad_interval code: 4 -> every 32 DWORDs (K rows),
// 3 -> every 16 DWORDs (V rows); pad_amount code 3 -> 4 DWORDs (= LDPAD halves).
static __device__ __forceinline__ void tdm_load_tile_2d(
    unsigned lds_addr, const void* gptr, unsigned dim0, unsigned dim1,
    unsigned stride0, unsigned pad_int, unsigned pad_amt) {
  const unsigned long long ga = (unsigned long long)(size_t)gptr;
  v4u g0;
  g0[0] = 1u;                                      // count=1 (valid), user mode
  g0[1] = lds_addr;                                // LDS byte address
  g0[2] = (unsigned)(ga & 0xFFFFFFFFu);            // global_addr[31:0]
  g0[3] = (unsigned)((ga >> 32) & 0x01FFFFFFu) | (2u << 30);  // [120:96], type=2
  v8i g1;
  g1[0] = (int)((1u << 16) |                       // data_size = 2 bytes
                (1u << 20) |                       // pad_enable
                (pad_int << 22) | (pad_amt << 25));
  g1[1] = (int)((dim0 & 0xFFFFu) << 16);           // tensor_dim0[15:0]
  g1[2] = (int)(((dim0 >> 16) & 0xFFFFu) | ((dim1 & 0xFFFFu) << 16));
  g1[3] = (int)(((dim1 >> 16) & 0xFFFFu) | ((dim0 & 0xFFFFu) << 16)); // tile_dim0
  g1[4] = (int)(dim1 & 0xFFFFu);                   // tile_dim1 (tile_dim2 = 0)
  g1[5] = (int)stride0;                            // tensor_dim0_stride[31:0]
  g1[6] = 0;
  g1[7] = 0;
  const v4i gz = {0, 0, 0, 0};
#if defined(__clang_major__) && (__clang_major__ >= 23)
  const v8i gz8 = {0, 0, 0, 0, 0, 0, 0, 0};
  __builtin_amdgcn_tensor_load_to_lds(g0, g1, gz, gz, gz8, 0);
#else
  __builtin_amdgcn_tensor_load_to_lds(g0, g1, gz, gz, 0);
#endif
}

// ---- GEMM + bias: out = A[4096x1024] * W[1024x1024] + bias ------------------
// MODE 0: bf16 out [B,H,S,64] (Q,K)   MODE 1: bf16 out [B,H,64,S] (V)
// MODE 2: f32  out [4096,1024] (final projection)
template <int MODE>
__global__ __launch_bounds__(256) void gemm_bias_kernel(
    const float* __restrict__ A, const float* __restrict__ W,
    const float* __restrict__ bias, void* __restrict__ outp) {
  __shared__ alignas(16) bf16_t As[BM][BK + LDPAD];
  __shared__ alignas(16) bf16_t Bs[BN][BK + LDPAD];

  const int t = threadIdx.x;
  const int rowBase = blockIdx.y * BM;
  const int colBase = blockIdx.x * BN;
  const int wave = t >> 5, ln = t & 31;
  const int wm = (wave & 3) * 32;        // 4 row-groups -> 128 rows
  const int wn = (wave >> 2) * 64;       // 2 col-groups -> 128 cols
  const int mrow = ln & 15, half = ln >> 4;

  const int ar = t >> 3;                 // A loader: 32 rows/pass, 4 floats along K
  const int ac = (t & 7) * 4;
  const int bk = t >> 5;                 // W loader: 8 k-rows/pass, 4 floats along N
  const int bn = (t & 31) * 4;

  v8f acc[2][4] = {};

  for (int k0 = 0; k0 < DMODEL; k0 += BK) {
    #pragma unroll
    for (int rr = 0; rr < BM; rr += 32) {        // A tile fp32 -> bf16
      const float4 f = *(const float4*)&A[(size_t)(rowBase + rr + ar) * DMODEL + k0 + ac];
      v4bf h8; h8[0] = (bf16_t)f.x; h8[1] = (bf16_t)f.y;
               h8[2] = (bf16_t)f.z; h8[3] = (bf16_t)f.w;
      *(v4bf*)&As[rr + ar][ac] = h8;
    }
    #pragma unroll
    for (int kk = 0; kk < BK; kk += 8) {         // W tile -> Bs[n][k] (transposed)
      const float4 f = *(const float4*)&W[(size_t)(k0 + kk + bk) * DMODEL + colBase + bn];
      Bs[bn + 0][kk + bk] = (bf16_t)f.x;
      Bs[bn + 1][kk + bk] = (bf16_t)f.y;
      Bs[bn + 2][kk + bk] = (bf16_t)f.z;
      Bs[bn + 3][kk + bk] = (bf16_t)f.w;
    }
    if (k0 + BK < DMODEL) {                      // global_prefetch_b8 next tiles
      __builtin_prefetch(&A[(size_t)(rowBase + ar) * DMODEL + k0 + BK + ac], 0, 1);
      __builtin_prefetch(&W[(size_t)(k0 + BK + bk) * DMODEL + colBase + bn], 0, 1);
    }
    __syncthreads();

    const v16bf a0 = load_a_frag(&As[wm + mrow][0], half);
    const v16bf a1 = load_a_frag(&As[wm + 16 + mrow][0], half);
    #pragma unroll
    for (int ni = 0; ni < 4; ++ni) {
      const v16bf b = load_b_frag(&Bs[wn + ni * 16 + mrow][0], half);
      acc[0][ni] = wmma_bf16(a0, b, acc[0][ni]);
      acc[1][ni] = wmma_bf16(a1, b, acc[1][ni]);
    }
    __syncthreads();
  }

  #pragma unroll
  for (int mi = 0; mi < 2; ++mi)
    #pragma unroll
    for (int ni = 0; ni < 4; ++ni) {
      const int col = colBase + wn + ni * 16 + mrow;
      const float bvv = bias[col];
      #pragma unroll
      for (int r = 0; r < 8; ++r) {
        const int row = rowBase + wm + mi * 16 + r + 8 * half;
        const float val = acc[mi][ni][r] + bvv;
        if (MODE == 2) {
          ((float*)outp)[(size_t)row * DMODEL + col] = val;
        } else {
          const int bb = row >> 11, ss = row & (SEQ - 1);
          const int hh = col >> 6,  dd = col & (DHEAD - 1);
          const size_t idx = (MODE == 0)
              ? ((size_t)(bb * NHEAD + hh) * SEQ + ss) * DHEAD + dd
              : ((size_t)(bb * NHEAD + hh) * DHEAD + dd) * SEQ + ss;
          ((bf16_t*)outp)[idx] = (bf16_t)val;
        }
      }
    }
}

// ---- Flash attention: 4 waves/WG, one 16-query tile per wave ----------------
// Q,K: bf16 [B,H,S,64]; Vt: bf16 [B,H,64,S]; ctx: f32 [B*S,1024] (heads merged)
// K/V tiles double-buffered in LDS via TDM, issued by wave 0.
__global__ __launch_bounds__(128) void flash_attn_kernel(
    const bf16_t* __restrict__ Q, const bf16_t* __restrict__ K,
    const bf16_t* __restrict__ Vt, const int* __restrict__ mask,
    float* __restrict__ ctx) {
  __shared__ alignas(16) bf16_t Ks[2][32][DHEAD + LDPAD];  // rows padded by TDM
  __shared__ alignas(16) bf16_t Vs[2][DHEAD][32 + LDPAD];
  __shared__ alignas(16) bf16_t Pb[4][16][32 + LDPAD];

  const int wave = threadIdx.x >> 5, ln = threadIdx.x & 31;
  const int mrow = ln & 15, half = ln >> 4;
  const int h = blockIdx.y, b = blockIdx.z;
  const int m0 = (blockIdx.x * 4 + wave) * 16;
  const size_t bh = (size_t)(b * NHEAD + h);
  const bf16_t* qp = Q + bh * SEQ * DHEAD;
  const bf16_t* kp = K + bh * SEQ * DHEAD;
  const bf16_t* vp = Vt + bh * DHEAD * SEQ;
  const int* mk = mask + b * SEQ;

  const v16bf qf0 = load_a_frag(qp + (size_t)(m0 + mrow) * DHEAD, half);
  const v16bf qf1 = load_a_frag(qp + (size_t)(m0 + mrow) * DHEAD + 32, half);

  v16bf onesf;
  #pragma unroll
  for (int i = 0; i < 16; ++i) onesf[i] = (bf16_t)1.0f;

  float mst[8];
  v8f o[4] = {};
  v8f lacc = {};                         // row sums via WMMA against ones
  #pragma unroll
  for (int r = 0; r < 8; ++r) mst[r] = -3.0e38f;

  if (wave == 0) {                       // prologue: DMA first K/V tiles
    tdm_load_tile_2d((unsigned)(size_t)&Ks[0][0][0], kp, DHEAD, 32, DHEAD, 4, 3);
    tdm_load_tile_2d((unsigned)(size_t)&Vs[0][0][0], vp, 32, DHEAD, SEQ, 3, 3);
  }

  for (int kt = 0; kt < NKT; ++kt) {
    const int buf = kt & 1;
    if (wave == 0) __builtin_amdgcn_s_wait_tensorcnt(0);
    __syncthreads();                     // tile kt ready; prev compute done
    if (wave == 0 && kt + 1 < NKT) {     // DMA tile kt+1 into other buffer
      tdm_load_tile_2d((unsigned)(size_t)&Ks[buf ^ 1][0][0],
                       kp + (size_t)(kt + 1) * 32 * DHEAD, DHEAD, 32, DHEAD, 4, 3);
      tdm_load_tile_2d((unsigned)(size_t)&Vs[buf ^ 1][0][0],
                       vp + (size_t)(kt + 1) * 32, 32, DHEAD, SEQ, 3, 3);
    }

    // ---- S = (Q Kt) * scale + mask : two 16x16 score tiles from LDS K ----
    v8f sc[2];
    #pragma unroll
    for (int tt = 0; tt < 2; ++tt) {
      v8f c = {};
      c = wmma_bf16(qf0, load_b_frag(&Ks[buf][tt * 16 + mrow][0], half), c);
      c = wmma_bf16(qf1, load_b_frag(&Ks[buf][tt * 16 + mrow][32], half), c);
      const float madd = (mk[kt * 32 + tt * 16 + mrow] != 0) ? -1.0e9f : 0.0f;
      #pragma unroll
      for (int r = 0; r < 8; ++r) c[r] = c[r] * 0.125f + madd;
      sc[tt] = c;
    }
    // ---- online softmax: row max across the 16 lanes of each half ----
    float alpha[8];
    #pragma unroll
    for (int r = 0; r < 8; ++r) {
      float v = fmaxf(sc[0][r], sc[1][r]);
      #pragma unroll
      for (int off = 1; off < 16; off <<= 1) v = fmaxf(v, __shfl_xor(v, off, 32));
      const float mnew = fmaxf(mst[r], v);
      alpha[r] = __expf(mst[r] - mnew);
      mst[r] = mnew;
    }
    #pragma unroll
    for (int tt = 0; tt < 2; ++tt)
      #pragma unroll
      for (int r = 0; r < 8; ++r) {
        const float p = __expf(sc[tt][r] - mst[r]);
        Pb[wave][r + 8 * half][tt * 16 + mrow] = (bf16_t)p;  // C-layout -> [M][K]
      }
    #pragma unroll
    for (int dt = 0; dt < 4; ++dt)
      #pragma unroll
      for (int r = 0; r < 8; ++r) o[dt][r] *= alpha[r];
    #pragma unroll
    for (int r = 0; r < 8; ++r) lacc[r] *= alpha[r];

    asm volatile("" ::: "memory");       // keep wave-private P stores before loads

    // ---- O += P x V (4 WMMA) and rowsum l += P x 1 (1 WMMA) ----
    const v16bf pf = load_a_frag(&Pb[wave][mrow][0], half);
    lacc = wmma_bf16(pf, onesf, lacc);
    #pragma unroll
    for (int dt = 0; dt < 4; ++dt) {
      const v16bf vf = load_b_frag(&Vs[buf][dt * 16 + mrow][0], half);
      o[dt] = wmma_bf16(pf, vf, o[dt]);
    }
  }

  // normalize and write ctx with heads merged: [b, s, h*64 + d]
  #pragma unroll
  for (int dt = 0; dt < 4; ++dt) {
    const int col = h * DHEAD + dt * 16 + mrow;
    #pragma unroll
    for (int r = 0; r < 8; ++r) {
      const int s = m0 + r + 8 * half;
      ctx[((size_t)b * SEQ + s) * DMODEL + col] = o[dt][r] / lacc[r];
    }
  }
}

// ---- launcher ---------------------------------------------------------------
extern "C" void kernel_launch(void* const* d_in, const int* in_sizes, int n_in,
                              void* d_out, int out_size, void* d_ws, size_t ws_size,
                              hipStream_t stream) {
  const float* query = (const float*)d_in[0];
  const float* key   = (const float*)d_in[1];
  const float* value = (const float*)d_in[2];
  const int*   mask  = (const int*)d_in[3];
  const float* Wq = (const float*)d_in[4];  const float* bq = (const float*)d_in[5];
  const float* Wk = (const float*)d_in[6];  const float* bk = (const float*)d_in[7];
  const float* Wv = (const float*)d_in[8];  const float* bv = (const float*)d_in[9];
  const float* Wo = (const float*)d_in[10]; const float* bo = (const float*)d_in[11];

  bf16_t* qb   = (bf16_t*)d_ws;                            //  8 MB [B,H,S,64]
  bf16_t* kbuf = qb   + (size_t)MROWS * DMODEL;            //  8 MB [B,H,S,64]
  bf16_t* vbuf = kbuf + (size_t)MROWS * DMODEL;            //  8 MB [B,H,64,S]
  float*  ctx  = (float*)(vbuf + (size_t)MROWS * DMODEL);  // 16 MB [4096,1024]

  const dim3 ggrid(DMODEL / BN, MROWS / BM);               // (8, 32)
  gemm_bias_kernel<0><<<ggrid, 256, 0, stream>>>(query, Wq, bq, qb);
  gemm_bias_kernel<0><<<ggrid, 256, 0, stream>>>(key,   Wk, bk, kbuf);
  gemm_bias_kernel<1><<<ggrid, 256, 0, stream>>>(value, Wv, bv, vbuf);
  flash_attn_kernel<<<dim3(SEQ / 16 / 4, NHEAD, BATCH), 128, 0, stream>>>(
      qb, kbuf, vbuf, mask, ctx);
  gemm_bias_kernel<2><<<ggrid, 256, 0, stream>>>(ctx, Wo, bo, (float*)d_out);
}
